// RelPosMultiHeadAttention_7541962572326
// MI455X (gfx1250) — matrix operand order repaired and checked by reading, not compiled, not run on hardware
//
#include <hip/hip_runtime.h>

typedef __attribute__((ext_vector_type(16))) _Float16 v16h;
typedef __attribute__((ext_vector_type(8)))  _Float16 v8h;
typedef __attribute__((ext_vector_type(8)))  float    v8f;

static constexpr int BB = 8, TT = 1024, DD = 512, HH = 8, DKK = 64, PP = 2047;
static constexpr int BT = BB * TT;

// ---------------- CDNA5 async memory->LDS helpers ----------------

// Generic pointer -> 32-bit LDS byte offset (addrspace(3) value).
__device__ inline unsigned lds_off(const _Float16* p) {
  return (unsigned)(size_t)(const __attribute__((address_space(3))) _Float16*)p;
}

// 16B per lane, memory -> LDS, tracked by ASYNCcnt (no VGPR data round-trip).
__device__ inline void async_copy_b128(unsigned lds, const void* gptr) {
  asm volatile("global_load_async_to_lds_b128 %0, %1, off"
               :: "v"(lds), "v"(gptr) : "memory");
}

__device__ inline void wait_async0() {
  asm volatile("s_wait_asynccnt 0x0" ::: "memory");
}

// ---------------- WMMA helpers (CDNA5 16x16x32 f16 -> f32) ----------------

// A-matrix fragment (M=16 x K=32): lanes 0-15 hold K 0-7 & 16-23; lanes 16-31
// hold K 8-15 & 24-31, for row M = lane%16 (per CDNA5 ISA A-layout table).
__device__ inline v16h frag_A(const _Float16* lds, int rowBase, int stride, int kBase) {
  int lane = threadIdx.x & 31;
  const _Float16* p = lds + (size_t)(rowBase + (lane & 15)) * stride + kBase + ((lane >> 4) << 3);
  v8h c0 = *(const v8h*)(p);
  v8h c1 = *(const v8h*)(p + 16);
  return __builtin_shufflevector(c0, c1, 0,1,2,3,4,5,6,7,8,9,10,11,12,13,14,15);
}

// B-matrix fragment (K=32 x N=16): lane holds column N=lane%16; lanes 0-15
// carry K 0-7 & 16-23, lanes 16-31 carry K 8-15 & 24-31. Source is B^T rows.
__device__ inline v16h frag_B(const _Float16* lds, int rowBase, int stride, int kBase) {
  int lane = threadIdx.x & 31;
  const _Float16* p = lds + (size_t)(rowBase + (lane & 15)) * stride + kBase + ((lane >> 4) << 3);
  v8h c0 = *(const v8h*)(p);
  v8h c1 = *(const v8h*)(p + 16);
  return __builtin_shufflevector(c0, c1, 0,1,2,3,4,5,6,7,8,9,10,11,12,13,14,15);
}

__device__ inline v8f wmma_f16(v16h a, v16h b, v8f c) {
  return __builtin_amdgcn_wmma_f32_16x16x32_f16(false, a, false, b, (short)0, c, false, false);
}

__device__ inline v8f zero8() {
  v8f z;
  for (int i = 0; i < 8; ++i) z[i] = 0.0f;
  return z;
}

// ---------------- LayerNorm (q/k/v selected by blockIdx.y) ----------------

__global__ __launch_bounds__(256) void ln_kernel(
    const float* __restrict__ q, const float* __restrict__ k, const float* __restrict__ v,
    const float* __restrict__ gq, const float* __restrict__ bq,
    const float* __restrict__ gk, const float* __restrict__ bk,
    const float* __restrict__ gv, const float* __restrict__ bv,
    _Float16* __restrict__ qn, _Float16* __restrict__ kn, _Float16* __restrict__ vn)
{
  __shared__ float s0[8], s1[8];
  int which = blockIdx.y;
  const float* in = (which == 0) ? q : (which == 1) ? k : v;
  const float* g  = (which == 0) ? gq : (which == 1) ? gk : gv;
  const float* be = (which == 0) ? bq : (which == 1) ? bk : bv;
  _Float16* out   = (which == 0) ? qn : (which == 1) ? kn : vn;

  size_t row = blockIdx.x;
  int tid = threadIdx.x;
  float x0 = in[row * DD + tid];
  float x1 = in[row * DD + tid + 256];
  float s = x0 + x1, ss = x0 * x0 + x1 * x1;
#pragma unroll
  for (int o = 16; o > 0; o >>= 1) { s += __shfl_xor(s, o, 32); ss += __shfl_xor(ss, o, 32); }
  if ((tid & 31) == 0) { s0[tid >> 5] = s; s1[tid >> 5] = ss; }
  __syncthreads();
  float ts = 0.f, tss = 0.f;
#pragma unroll
  for (int i = 0; i < 8; ++i) { ts += s0[i]; tss += s1[i]; }
  float mu = ts * (1.0f / DD);
  float var = tss * (1.0f / DD) - mu * mu;
  float rstd = rsqrtf(var + 1e-5f);
  out[row * DD + tid]       = (_Float16)((x0 - mu) * rstd * g[tid]       + be[tid]);
  out[row * DD + tid + 256] = (_Float16)((x1 - mu) * rstd * g[tid + 256] + be[tid + 256]);
}

// ---------------- f32 -> f16 convert ----------------

__global__ __launch_bounds__(256) void cvt_f16(const float* __restrict__ src,
                                               _Float16* __restrict__ dst, int n)
{
  int i = blockIdx.x * 256 + threadIdx.x;
  if (i < n) dst[i] = (_Float16)src[i];
}

// ---------------- Generic K=512 WMMA GEMM: C = A * W^T, mode-select epilogue
// mode 0: q_u/q_v = C + bias (write [B,H,T,dk] f16 x2)
// mode 1: kh [B,H,T,dk] f16
// mode 2: vhT [B,H,dk,T] f16 (transposed)
// mode 3: p_h [H,P,dk] f16 (M=2047, guarded)
// mode 4: out f32 = C + residual
__global__ __launch_bounds__(128) void gemm_proj(
    const _Float16* __restrict__ A, const _Float16* __restrict__ W,
    int Mvalid, int mode,
    _Float16* __restrict__ out0, _Float16* __restrict__ out1,
    const float* __restrict__ aux0, const float* __restrict__ aux1,
    float* __restrict__ outF)
{
  __shared__ _Float16 As[64 * 32];
  __shared__ _Float16 Bs[64 * 32];
  int tid = threadIdx.x;
  int lane = tid & 31, wv = tid >> 5;
  int m0 = blockIdx.x * 64;
  int n0 = blockIdx.y * 64;
  int mclamp = Mvalid - 1;

  v8f acc[4];
#pragma unroll
  for (int n = 0; n < 4; ++n) acc[n] = zero8();

  for (int k0 = 0; k0 < DD; k0 += 32) {
    __syncthreads();
#pragma unroll
    for (int rr = 0; rr < 2; ++rr) {
      int hidx = rr * 1024 + tid * 8;
      int row = hidx >> 5, cc = hidx & 31;
      int m = m0 + row; if (m > mclamp) m = mclamp;
      async_copy_b128(lds_off(As + hidx), A + (size_t)m * DD + k0 + cc);
      async_copy_b128(lds_off(Bs + hidx), W + (size_t)(n0 + row) * DD + k0 + cc);
    }
    wait_async0();
    __syncthreads();
    v16h a = frag_A(As, 16 * wv, 32, 0);
#pragma unroll
    for (int n = 0; n < 4; ++n)
      acc[n] = wmma_f16(a, frag_B(Bs, 16 * n, 32, 0), acc[n]);
  }

#pragma unroll
  for (int n = 0; n < 4; ++n) {
#pragma unroll
    for (int r = 0; r < 8; ++r) {
      int m = m0 + 16 * wv + r + 8 * (lane >> 4);
      int o = n0 + 16 * n + (lane & 15);
      float val = acc[n][r];
      if (mode == 0) {
        int b = m >> 10, t = m & 1023, h = o >> 6, d = o & 63;
        size_t idx = (((size_t)(b * HH + h)) * TT + t) * DKK + d;
        out0[idx] = (_Float16)(val + aux0[o]);
        out1[idx] = (_Float16)(val + aux1[o]);
      } else if (mode == 1) {
        int b = m >> 10, t = m & 1023, h = o >> 6, d = o & 63;
        out0[(((size_t)(b * HH + h)) * TT + t) * DKK + d] = (_Float16)val;
      } else if (mode == 2) {
        int b = m >> 10, t = m & 1023, h = o >> 6, d = o & 63;
        out0[(((size_t)(b * HH + h)) * DKK + d) * TT + t] = (_Float16)val;
      } else if (mode == 3) {
        if (m < Mvalid) {
          int h = o >> 6, d = o & 63;
          out0[((size_t)h * PP + m) * DKK + d] = (_Float16)val;
        }
      } else {
        outF[(size_t)m * DD + o] = val + aux0[(size_t)m * DD + o];
      }
    }
  }
}

// ---------------- Scores: AC + rel-shifted BD, scaled, f32 out ----------------
// scores[bh,q,k] = (q_u[q,:]*kh[k,:] + q_v[q,:]*p[(T-1)+k-q,:]) * scale

__global__ __launch_bounds__(128) void scores_kernel(
    const _Float16* __restrict__ qu, const _Float16* __restrict__ qv,
    const _Float16* __restrict__ kh, const _Float16* __restrict__ ph,
    float* __restrict__ sc)
{
  __shared__ _Float16 QU[64 * 64], QV[64 * 64], KHs[64 * 64], PS[128 * 64];
  int tid = threadIdx.x, lane = tid & 31, wv = tid >> 5;
  int bh = blockIdx.x;
  int h = bh & 7;
  int q0 = blockIdx.y * 64, k0 = blockIdx.z * 64;

  const _Float16* quB = qu + ((size_t)bh * TT + q0) * DKK;
  const _Float16* qvB = qv + ((size_t)bh * TT + q0) * DKK;
  const _Float16* khB = kh + ((size_t)bh * TT + k0) * DKK;
  int jslab0 = TT - 1 + k0 - q0 - 63;   // always >= 0 for T=1024 tiles

#pragma unroll
  for (int rr = 0; rr < 4; ++rr) {
    int hidx = rr * 1024 + tid * 8;
    async_copy_b128(lds_off(QU  + hidx), quB + hidx);
    async_copy_b128(lds_off(QV  + hidx), qvB + hidx);
    async_copy_b128(lds_off(KHs + hidx), khB + hidx);
  }
#pragma unroll
  for (int rr = 0; rr < 8; ++rr) {
    int hidx = rr * 1024 + tid * 8;
    int row = hidx >> 6, cc = hidx & 63;
    int j = jslab0 + row; if (j > PP - 1) j = PP - 1;   // row 127 may be 1 past band
    async_copy_b128(lds_off(PS + hidx), ph + ((size_t)h * PP + j) * DKK + cc);
  }
  wait_async0();
  __syncthreads();

  v16h a0  = frag_A(QU, 16 * wv, 64, 0);
  v16h a1  = frag_A(QU, 16 * wv, 64, 32);
  v16h av0 = frag_A(QV, 16 * wv, 64, 0);
  v16h av1 = frag_A(QV, 16 * wv, 64, 32);
  const float scale = 0.125f;   // 1/sqrt(64)

  for (int n = 0; n < 4; ++n) {
    v8f acc = zero8();
    acc = wmma_f16(a0, frag_B(KHs, 16 * n, 64, 0),  acc);
    acc = wmma_f16(a1, frag_B(KHs, 16 * n, 64, 32), acc);

    int pbase = 16 * n - 16 * wv + 48;   // local slab row for c=0
    v8f pA = zero8(), pB = zero8();
    pA = wmma_f16(av0, frag_B(PS, pbase,      64, 0),  pA);
    pA = wmma_f16(av1, frag_B(PS, pbase,      64, 32), pA);
    pB = wmma_f16(av0, frag_B(PS, pbase + 16, 64, 0),  pB);
    pB = wmma_f16(av1, frag_B(PS, pbase + 16, 64, 32), pB);

    int kcol = k0 + 16 * n + (lane & 15);
    int hilo = (lane >> 4) << 4;
#pragma unroll
    for (int r = 0; r < 8; ++r) {
      int qq = r + 8 * (lane >> 4);
      int c = 15 + (lane & 15) - qq;     // 0..30: diagonal band select
      float fa = __shfl(pA[r], (c & 15) + hilo, 32);
      float fb = __shfl(pB[r], ((c - 16) & 15) + hilo, 32);
      float bd = (c < 16) ? fa : fb;
      int qrow = q0 + 16 * wv + qq;
      sc[((size_t)bh * TT + qrow) * TT + kcol] = (acc[r] + bd) * scale;
    }
  }
}

// ---------------- Masked softmax, in place over rows of 1024 ----------------

__global__ __launch_bounds__(256) void softmax_kernel(float* __restrict__ sc,
                                                      const int* __restrict__ mask)
{
  __shared__ float red[8];
  int row = blockIdx.x;                 // bh*T + q
  int bh = row >> 10, qq = row & 1023;
  int b = bh >> 3;
  float* s = sc + (size_t)row * TT;
  const int* mrow = mask + ((size_t)b * TT + qq) * TT;
  int tid = threadIdx.x;

  float vals[4]; int ms[4];
  float mx = -3.4e38f;
#pragma unroll
  for (int i = 0; i < 4; ++i) {
    int c = tid + i * 256;
    ms[i] = mrow[c];
    float x = s[c];
    vals[i] = (ms[i] == 0) ? -3.4e38f : x;
    mx = fmaxf(mx, vals[i]);
  }
#pragma unroll
  for (int o = 16; o > 0; o >>= 1) mx = fmaxf(mx, __shfl_xor(mx, o, 32));
  if ((tid & 31) == 0) red[tid >> 5] = mx;
  __syncthreads();
  float m2 = red[0];
#pragma unroll
  for (int i = 1; i < 8; ++i) m2 = fmaxf(m2, red[i]);
  __syncthreads();

  float sum = 0.f;
#pragma unroll
  for (int i = 0; i < 4; ++i) {
    float e = ms[i] ? __expf(vals[i] - m2) : 0.0f;
    vals[i] = e;
    sum += e;
  }
#pragma unroll
  for (int o = 16; o > 0; o >>= 1) sum += __shfl_xor(sum, o, 32);
  if ((tid & 31) == 0) red[tid >> 5] = sum;
  __syncthreads();
  float tot = 0.f;
#pragma unroll
  for (int i = 0; i < 8; ++i) tot += red[i];
  float inv = 1.0f / tot;
#pragma unroll
  for (int i = 0; i < 4; ++i) s[tid + i * 256] = vals[i] * inv;
}

// ---------------- ctx = attn @ vh  (f32 attn -> f16 on the fly) ----------------

__global__ __launch_bounds__(128) void attn_v(
    const float* __restrict__ attn, const _Float16* __restrict__ vt,
    _Float16* __restrict__ ctx)
{
  __shared__ _Float16 As[64 * 32], Vs[64 * 32];
  int tid = threadIdx.x, lane = tid & 31, wv = tid >> 5;
  int bh = blockIdx.x; int b = bh >> 3, h = bh & 7;
  int q0 = blockIdx.y * 64;

  v8f acc[4];
#pragma unroll
  for (int n = 0; n < 4; ++n) acc[n] = zero8();

  for (int kk0 = 0; kk0 < TT; kk0 += 32) {
    __syncthreads();
    // V tile: pure copy -> async memory->LDS path (ASYNCcnt)
#pragma unroll
    for (int rr = 0; rr < 2; ++rr) {
      int hidx = rr * 1024 + tid * 8;
      int row = hidx >> 5, cc = hidx & 31;
      async_copy_b128(lds_off(Vs + hidx), vt + ((size_t)bh * DKK + row) * TT + kk0 + cc);
    }
    // attn tile needs f32->f16 convert, so it stays on the VGPR path
#pragma unroll
    for (int rr = 0; rr < 4; ++rr) {
      int eidx = rr * 512 + tid * 4;
      int row = eidx >> 5, cc = eidx & 31;
      float4 f = *(const float4*)(attn + ((size_t)bh * TT + q0 + row) * TT + kk0 + cc);
      _Float16* d = As + eidx;
      d[0] = (_Float16)f.x; d[1] = (_Float16)f.y; d[2] = (_Float16)f.z; d[3] = (_Float16)f.w;
    }
    wait_async0();
    __syncthreads();
    v16h a = frag_A(As, 16 * wv, 32, 0);
#pragma unroll
    for (int n = 0; n < 4; ++n)
      acc[n] = wmma_f16(a, frag_B(Vs, 16 * n, 32, 0), acc[n]);
  }

#pragma unroll
  for (int n = 0; n < 4; ++n) {
#pragma unroll
    for (int r = 0; r < 8; ++r) {
      int t = q0 + 16 * wv + r + 8 * (lane >> 4);
      int o = 16 * n + (lane & 15);
      ctx[((size_t)b * TT + t) * DD + h * DKK + o] = (_Float16)acc[n][r];
    }
  }
}

// ---------------- host side ----------------

extern "C" void kernel_launch(void* const* d_in, const int* in_sizes, int n_in,
                              void* d_out, int out_size, void* d_ws, size_t ws_size,
                              hipStream_t stream)
{
  (void)in_sizes; (void)n_in; (void)out_size; (void)ws_size;
  const float* q    = (const float*)d_in[0];
  const float* k    = (const float*)d_in[1];
  const float* v    = (const float*)d_in[2];
  const float* pe   = (const float*)d_in[3];
  const int*   mask = (const int*)  d_in[4];
  const float* gq = (const float*)d_in[5],  *bq = (const float*)d_in[6];
  const float* gk = (const float*)d_in[7],  *bk = (const float*)d_in[8];
  const float* gv = (const float*)d_in[9],  *bv = (const float*)d_in[10];
  const float* wq = (const float*)d_in[11], *wk = (const float*)d_in[12];
  const float* wvw = (const float*)d_in[13], *wpos = (const float*)d_in[14];
  const float* wfc = (const float*)d_in[15];
  const float* bias_u = (const float*)d_in[16], *bias_v = (const float*)d_in[17];

  char* ws = (char*)d_ws;
  size_t off = 0;
  auto alloc = [&](size_t bytes) -> char* {
    char* p = ws + off;
    off = (off + bytes + 255) & ~(size_t)255;
    return p;
  };
  _Float16* qn   = (_Float16*)alloc((size_t)BT * DD * 2);
  _Float16* kn   = (_Float16*)alloc((size_t)BT * DD * 2);
  _Float16* vn   = (_Float16*)alloc((size_t)BT * DD * 2);
  _Float16* posn = (_Float16*)alloc((size_t)PP * DD * 2);
  _Float16* wqh  = (_Float16*)alloc((size_t)DD * DD * 2);
  _Float16* wkh  = (_Float16*)alloc((size_t)DD * DD * 2);
  _Float16* wvh  = (_Float16*)alloc((size_t)DD * DD * 2);
  _Float16* wph  = (_Float16*)alloc((size_t)DD * DD * 2);
  _Float16* wfh  = (_Float16*)alloc((size_t)DD * DD * 2);
  _Float16* quW  = (_Float16*)alloc((size_t)BT * DD * 2);
  _Float16* qvW  = (_Float16*)alloc((size_t)BT * DD * 2);
  _Float16* khW  = (_Float16*)alloc((size_t)BT * DD * 2);
  _Float16* vtW  = (_Float16*)alloc((size_t)BT * DD * 2);
  _Float16* phW  = (_Float16*)alloc((size_t)HH * PP * DKK * 2);
  _Float16* ctx  = (_Float16*)alloc((size_t)BT * DD * 2);

  float* outO = (float*)d_out;
  float* attn = outO + (size_t)BT * DD;   // d_out = [out | attn]

  ln_kernel<<<dim3(BT, 3), 256, 0, stream>>>(q, k, v, gq, bq, gk, bk, gv, bv, qn, kn, vn);

  cvt_f16<<<(DD * DD + 255) / 256, 256, 0, stream>>>(wq,   wqh, DD * DD);
  cvt_f16<<<(DD * DD + 255) / 256, 256, 0, stream>>>(wk,   wkh, DD * DD);
  cvt_f16<<<(DD * DD + 255) / 256, 256, 0, stream>>>(wvw,  wvh, DD * DD);
  cvt_f16<<<(DD * DD + 255) / 256, 256, 0, stream>>>(wpos, wph, DD * DD);
  cvt_f16<<<(DD * DD + 255) / 256, 256, 0, stream>>>(wfc,  wfh, DD * DD);
  cvt_f16<<<(PP * DD + 255) / 256, 256, 0, stream>>>(pe,   posn, PP * DD);

  dim3 gemmGrid(BT / 64, DD / 64);
  gemm_proj<<<gemmGrid, 128, 0, stream>>>(qn, wqh, BT, 0, quW, qvW, bias_u, bias_v, nullptr);
  gemm_proj<<<gemmGrid, 128, 0, stream>>>(kn, wkh, BT, 1, khW, nullptr, nullptr, nullptr, nullptr);
  gemm_proj<<<gemmGrid, 128, 0, stream>>>(vn, wvh, BT, 2, vtW, nullptr, nullptr, nullptr, nullptr);
  gemm_proj<<<dim3((PP + 63) / 64, DD / 64), 128, 0, stream>>>(posn, wph, PP, 3, phW, nullptr, nullptr, nullptr, nullptr);

  scores_kernel<<<dim3(BB * HH, TT / 64, TT / 64), 128, 0, stream>>>(quW, qvW, khW, phW, attn);
  softmax_kernel<<<BB * HH * TT, 256, 0, stream>>>(attn, mask);
  attn_v<<<dim3(BB * HH, TT / 64), 128, 0, stream>>>(attn, vtW, ctx);
  gemm_proj<<<gemmGrid, 128, 0, stream>>>(ctx, wfh, BT, 4, nullptr, nullptr, q, nullptr, outO);
}
